// VisualAttentionProcess_37142877176520
// MI455X (gfx1250) — matrix operand order repaired and checked by reading, not compiled, not run on hardware
//
#include <hip/hip_runtime.h>
#include <hip/hip_bf16.h>

typedef __bf16 bf16_t;
typedef __attribute__((ext_vector_type(16))) __bf16 v16bf;
typedef __attribute__((ext_vector_type(8)))  __bf16 v8bf;
typedef __attribute__((ext_vector_type(8)))  float  v8f;

#define NHEAD 20
#define DHEAD 64
#define SQ    4096
#define SKV   77
#define SKV_P 96
#define CIN   1280
#define CENC  2048
#define NB    2
#define MPAD_E 192   // encoder rows padded for 64-row block tiles

__device__ __forceinline__ v16bf combine16(v8bf lo, v8bf hi) {
    return __builtin_shufflevector(lo, hi, 0,1,2,3,4,5,6,7,8,9,10,11,12,13,14,15);
}

// A-fragment: 16x32 bf16, lane&15 = row, interleaved K-chunks of 8 per half-wave
__device__ __forceinline__ v16bf load_fragA(const bf16_t* base, int stride, int row0, int k0, int lane) {
    int r   = row0 + (lane & 15);
    int hi8 = (lane >> 4) << 3;
    const bf16_t* p = base + (long)r * stride + k0 + hi8;
    v8bf lo = *reinterpret_cast<const v8bf*>(p);
    v8bf hh = *reinterpret_cast<const v8bf*>(p + 16);
    return combine16(lo, hh);
}

// B-fragment from B^T [N][K]: lane&15 = col, interleaved K-chunks of 8 per half-wave
__device__ __forceinline__ v16bf load_fragB(const bf16_t* base, int stride, int col0, int k0, int lane) {
    int c    = col0 + (lane & 15);
    int hi16 = (lane >> 4) << 3;
    const bf16_t* p = base + (long)c * stride + k0 + hi16;
    v8bf lo = *reinterpret_cast<const v8bf*>(p);
    v8bf hh = *reinterpret_cast<const v8bf*>(p + 16);
    return combine16(lo, hh);
}

__device__ __forceinline__ v8f wmma_bf16(v16bf a, v16bf b, v8f c) {
    return __builtin_amdgcn_wmma_f32_16x16x32_bf16(false, a, false, b, (short)0, c, false, false);
}

// ---- CDNA5 async global->LDS copy (ASYNCcnt path) ----
__device__ __forceinline__ void async_copy16(unsigned int lds_off, const bf16_t* g) {
    asm volatile("global_load_async_to_lds_b128 %0, %1, off"
                 :: "v"(lds_off), "v"((unsigned long long)(uintptr_t)g)
                 : "memory");
}
__device__ __forceinline__ void wait_async0() {
    asm volatile("s_wait_asynccnt 0" ::: "memory");
}

// ---------------- conversion kernels ----------------
__global__ void cvt_bf16_kernel(const float* __restrict__ in, bf16_t* __restrict__ out,
                                long n_valid, long n_total) {
    long i = (long)blockIdx.x * blockDim.x + threadIdx.x;
    if (i < n_total) out[i] = (i < n_valid) ? (bf16_t)in[i] : (bf16_t)0.0f;
}

__global__ void cvt_transpose_kernel(const float* __restrict__ in, bf16_t* __restrict__ out,
                                     int K, int N) {
    long i = (long)blockIdx.x * blockDim.x + threadIdx.x;
    if (i < (long)K * N) {
        int k = (int)(i / N);
        int n = (int)(i % N);
        out[(long)n * K + k] = (bf16_t)in[i];
    }
}

// ---------------- generic bf16 GEMM, fp32 accum ----------------
// C[M,N] = A[M,K] * B^T[N,K] (+bias). Block = 256 thr (8 waves, 2Mx4N), wave tile 32x64.
__global__ void gemm_bf16_kernel(const bf16_t* __restrict__ A, const bf16_t* __restrict__ BT,
                                 const float* __restrict__ bias,
                                 float* __restrict__ outF, bf16_t* __restrict__ outB,
                                 int M, int N, int K, int Mreal) {
    int lane = threadIdx.x & 31;
    int w    = threadIdx.x >> 5;
    int wm   = w >> 2, wn = w & 3;
    int row0 = blockIdx.y * 64 + wm * 32;
    int n0   = blockIdx.x * 256 + wn * 64;

    v8f acc[2][4] = {};
    for (int k0 = 0; k0 < K; k0 += 32) {
        v16bf a0 = load_fragA(A, K, row0,      k0, lane);
        v16bf a1 = load_fragA(A, K, row0 + 16, k0, lane);
#pragma unroll
        for (int i = 0; i < 4; ++i) {
            v16bf bfr = load_fragB(BT, K, n0 + 16 * i, k0, lane);
            acc[0][i] = wmma_bf16(a0, bfr, acc[0][i]);
            acc[1][i] = wmma_bf16(a1, bfr, acc[1][i]);
        }
    }

    int col   = lane & 15;
    int hioff = (lane >> 4) << 3;
#pragma unroll
    for (int mi = 0; mi < 2; ++mi) {
        int rbase = row0 + mi * 16 + hioff;
#pragma unroll
        for (int i = 0; i < 4; ++i) {
            int n = n0 + 16 * i + col;
            float bv = bias ? bias[n] : 0.0f;
#pragma unroll
            for (int r = 0; r < 8; ++r) {
                int m = rbase + r;
                if (m < Mreal) {
                    float v = acc[mi][i][r] + bv;
                    if (outB) outB[(long)m * N + n] = (bf16_t)v;
                    else      outF[(long)m * N + n] = v;
                }
            }
        }
    }
}

// ---------------- CORA erase on V, writes V_new transposed per head ----------------
// Vt[b][head][d][s] (s padded to 96 with zeros), bf16.
__global__ void cora_kernel(const float* __restrict__ V, const float* __restrict__ Bpad,
                            const float* __restrict__ uhat, const float* __restrict__ ahat,
                            bf16_t* __restrict__ Vt) {
    int b = blockIdx.x / SKV_P;
    int s = blockIdx.x % SKV_P;
    int t = threadIdx.x;           // 256 threads, 5 channels each

    if (s >= SKV) {
        for (int ch = t; ch < CIN; ch += 256) {
            int head = ch >> 6, d = ch & 63;
            Vt[((long)(b * NHEAD + head) * DHEAD + d) * SKV_P + s] = (bf16_t)0.0f;
        }
        return;
    }

    __shared__ float red[4][256];

    float v[5], bp[5][4], uh[5], ah[5];
#pragma unroll
    for (int i = 0; i < 5; ++i) {
        int ch = t + i * 256;
        v[i]  = V[((long)b * SKV + s) * CIN + ch];
#pragma unroll
        for (int k = 0; k < 4; ++k) bp[i][k] = Bpad[((long)s * CIN + ch) * 4 + k];
        uh[i] = uhat[(long)s * CIN + ch];
        ah[i] = ahat[(long)s * CIN + ch];
    }

    float part[4] = {0.f, 0.f, 0.f, 0.f};
#pragma unroll
    for (int i = 0; i < 5; ++i)
#pragma unroll
        for (int k = 0; k < 4; ++k) part[k] += v[i] * bp[i][k];
#pragma unroll
    for (int k = 0; k < 4; ++k) red[k][t] = part[k];
    __syncthreads();
    for (int off = 128; off > 0; off >>= 1) {
        if (t < off)
#pragma unroll
            for (int k = 0; k < 4; ++k) red[k][t] += red[k][t + off];
        __syncthreads();
    }
    float coef[4];
#pragma unroll
    for (int k = 0; k < 4; ++k) coef[k] = red[k][0];
    __syncthreads();

    float vfree[5], vpres[5];
    float tp = 0.f, np = 0.f;
#pragma unroll
    for (int i = 0; i < 5; ++i) {
        float p = 0.f;
#pragma unroll
        for (int k = 0; k < 4; ++k) p += coef[k] * bp[i][k];
        vpres[i] = p;
        vfree[i] = v[i] - p;
        tp += vfree[i] * uh[i];
        np += vfree[i] * vfree[i];
    }
    red[0][t] = tp; red[1][t] = np;
    __syncthreads();
    for (int off = 128; off > 0; off >>= 1) {
        if (t < off) { red[0][t] += red[0][t + off]; red[1][t] += red[1][t + off]; }
        __syncthreads();
    }
    float tv    = red[0][0];
    float denom = sqrtf(red[1][0]) + 1e-8f;
    bool  gate  = (fabsf(tv) / denom) >= 0.1f;

#pragma unroll
    for (int i = 0; i < 5; ++i) {
        float vf  = vfree[i];
        float vfo = gate ? (vf - tv * uh[i] + 0.5f * tv * ah[i]) : vf;
        float vn  = vpres[i] + vfo;
        int ch = t + i * 256;
        int head = ch >> 6, d = ch & 63;
        Vt[((long)(b * NHEAD + head) * DHEAD + d) * SKV_P + s] = (bf16_t)vn;
    }
}

// ---------------- fused attention: scores -> softmax -> PV ----------------
// block = 128 thr (4 waves), each wave = 16 queries of one (b,h); grid = (SQ/64, B*H).
// K-slice and V^T-slice staged in LDS via async global->LDS copies (ASYNCcnt).
__global__ void attention_kernel(const bf16_t* __restrict__ Qb, const bf16_t* __restrict__ Kb,
                                 const bf16_t* __restrict__ Vt, bf16_t* __restrict__ Ob) {
    __shared__ __attribute__((aligned(16))) float  s_scores[4][16][80];
    __shared__ __attribute__((aligned(16))) bf16_t s_probs[4][16][SKV_P];
    __shared__ __attribute__((aligned(16))) bf16_t s_ktile[80][DHEAD];     // rows 77..79 garbage, unread cols
    __shared__ __attribute__((aligned(16))) bf16_t s_vtile[DHEAD][SKV_P];

    int bh = blockIdx.y;
    int b  = bh / NHEAD;
    int h  = bh % NHEAD;
    int lane = threadIdx.x & 31;
    int w    = threadIdx.x >> 5;
    int q0   = blockIdx.x * 64 + w * 16;

    int colc = lane & 15;
    int hi8  = (lane >> 4) << 3;
    const float scale = 0.125f;   // 1/sqrt(64)

    // ---- stage K-slice [77][64] and V^T-slice [64][96] into LDS asynchronously
    {
        unsigned int kbase = (unsigned int)(uintptr_t)(&s_ktile[0][0]);
        const bf16_t* ksrc = Kb + (long)b * SKV * CIN + h * DHEAD;
        for (int idx = threadIdx.x; idx < SKV * 8; idx += 128) {   // 16B chunks: 8 per row
            int kv = idx >> 3, c = idx & 7;
            async_copy16(kbase + (unsigned int)(kv * DHEAD + c * 8) * 2,
                         ksrc + (long)kv * CIN + c * 8);
        }
        unsigned int vbase = (unsigned int)(uintptr_t)(&s_vtile[0][0]);
        const bf16_t* vsrc = Vt + (long)(b * NHEAD + h) * DHEAD * SKV_P;
        for (int idx = threadIdx.x; idx < (DHEAD * SKV_P) / 8; idx += 128) {
            async_copy16(vbase + (unsigned int)idx * 16, vsrc + idx * 8);
        }
    }

    // ---- Q fragments from global (streamed once)
    const bf16_t* qbase = Qb + ((long)b * SQ + q0) * CIN + h * DHEAD;
    v16bf qa0 = load_fragA(qbase, CIN, 0, 0, lane);
    v16bf qa1 = load_fragA(qbase, CIN, 0, 32, lane);

    wait_async0();
    __syncthreads();

    // ---- Phase 1: S = Q K^T * scale (5 kv-tiles of 16, K-dim = 64 = 2 chunks)
    const bf16_t* kt = &s_ktile[0][0];
#pragma unroll
    for (int nt = 0; nt < 5; ++nt) {
        v16bf kb0 = load_fragB(kt, DHEAD, nt * 16, 0, lane);
        v16bf kb1 = load_fragB(kt, DHEAD, nt * 16, 32, lane);
        v8f acc = {};
        acc = wmma_bf16(qa0, kb0, acc);
        acc = wmma_bf16(qa1, kb1, acc);
#pragma unroll
        for (int r = 0; r < 8; ++r)
            s_scores[w][r + hi8][nt * 16 + colc] = acc[r] * scale;
    }
    __syncthreads();

    // ---- Phase 2: softmax over 77 kv (one thread per query row)
    if (threadIdx.x < 64) {
        int sw = threadIdx.x >> 4, sq = threadIdx.x & 15;
        float mx = -1e30f;
        for (int i = 0; i < SKV; ++i) mx = fmaxf(mx, s_scores[sw][sq][i]);
        float sum = 0.f;
        for (int i = 0; i < SKV; ++i) sum += __expf(s_scores[sw][sq][i] - mx);
        float inv = 1.0f / sum;
        for (int i = 0; i < SKV_P; ++i)
            s_probs[sw][sq][i] = (i < SKV) ? (bf16_t)(__expf(s_scores[sw][sq][i] - mx) * inv)
                                           : (bf16_t)0.0f;
    }
    __syncthreads();

    // ---- Phase 3: O = P @ V_new  (K-dim = 96 = 3 chunks, 4 d-tiles of 16)
    const bf16_t* pbase = &s_probs[w][0][0];
    v16bf pf[3];
#pragma unroll
    for (int kc = 0; kc < 3; ++kc) pf[kc] = load_fragA(pbase, SKV_P, 0, kc * 32, lane);

    const bf16_t* vtl = &s_vtile[0][0];
#pragma unroll
    for (int dt = 0; dt < 4; ++dt) {
        v8f acc = {};
#pragma unroll
        for (int kc = 0; kc < 3; ++kc) {
            v16bf vfrag = load_fragB(vtl, SKV_P, dt * 16, kc * 32, lane);
            acc = wmma_bf16(pf[kc], vfrag, acc);
        }
#pragma unroll
        for (int r = 0; r < 8; ++r) {
            long m = (long)b * SQ + q0 + r + hi8;
            Ob[m * CIN + h * DHEAD + dt * 16 + colc] = (bf16_t)acc[r];
        }
    }
}

// ---------------- host launch ----------------
extern "C" void kernel_launch(void* const* d_in, const int* in_sizes, int n_in,
                              void* d_out, int out_size, void* d_ws, size_t ws_size,
                              hipStream_t stream) {
    const float* hid  = (const float*)d_in[0];
    const float* enc  = (const float*)d_in[1];
    const float* Wq   = (const float*)d_in[2];
    const float* Wk   = (const float*)d_in[3];
    const float* Wv   = (const float*)d_in[4];
    const float* Wo   = (const float*)d_in[5];
    const float* bo   = (const float*)d_in[6];
    const float* Bpad = (const float*)d_in[7];
    const float* uhat = (const float*)d_in[8];
    const float* ahat = (const float*)d_in[9];
    float* out = (float*)d_out;

    char* ws = (char*)d_ws;
    size_t off = 0;
    auto take = [&](size_t bytes) -> char* {
        char* p = ws + off;
        off += (bytes + 255) & ~(size_t)255;
        return p;
    };

    bf16_t* Xh  = (bf16_t*)take((size_t)NB * SQ * CIN * 2);     // hidden bf16, reused as attn-out
    bf16_t* Xe  = (bf16_t*)take((size_t)MPAD_E * CENC * 2);     // encoder bf16, rows padded to 192
    bf16_t* WqT = (bf16_t*)take((size_t)CIN * CIN * 2);
    bf16_t* WkT = (bf16_t*)take((size_t)CIN * CENC * 2);
    bf16_t* WvT = (bf16_t*)take((size_t)CIN * CENC * 2);
    bf16_t* WoT = (bf16_t*)take((size_t)CIN * CIN * 2);
    bf16_t* Qb  = (bf16_t*)take((size_t)NB * SQ * CIN * 2);
    bf16_t* Kb  = (bf16_t*)take((size_t)NB * SKV * CIN * 2);
    float*  Vf  = (float*) take((size_t)NB * SKV * CIN * 4);
    bf16_t* Vt  = (bf16_t*)take((size_t)NB * NHEAD * DHEAD * SKV_P * 2);
    bf16_t* Attn = Xh;  // safe reuse: Xh fully consumed by Q-projection before attention runs

    // conversions
    {
        long n = (long)NB * SQ * CIN;
        cvt_bf16_kernel<<<dim3((unsigned)((n + 255) / 256)), 256, 0, stream>>>(hid, Xh, n, n);
    }
    {
        long nv = (long)NB * SKV * CENC, nt = (long)MPAD_E * CENC;
        cvt_bf16_kernel<<<dim3((unsigned)((nt + 255) / 256)), 256, 0, stream>>>(enc, Xe, nv, nt);
    }
    cvt_transpose_kernel<<<dim3((CIN * CIN + 255) / 256), 256, 0, stream>>>(Wq, WqT, CIN, CIN);
    cvt_transpose_kernel<<<dim3((CENC * CIN + 255) / 256), 256, 0, stream>>>(Wk, WkT, CENC, CIN);
    cvt_transpose_kernel<<<dim3((CENC * CIN + 255) / 256), 256, 0, stream>>>(Wv, WvT, CENC, CIN);
    cvt_transpose_kernel<<<dim3((CIN * CIN + 255) / 256), 256, 0, stream>>>(Wo, WoT, CIN, CIN);

    // Q = Xh @ Wq  -> bf16
    gemm_bf16_kernel<<<dim3(CIN / 256, (NB * SQ) / 64), 256, 0, stream>>>(
        Xh, WqT, nullptr, nullptr, Qb, NB * SQ, CIN, CIN, NB * SQ);
    // K = Xe @ Wk -> bf16   (M padded 192, real 154)
    gemm_bf16_kernel<<<dim3(CIN / 256, MPAD_E / 64), 256, 0, stream>>>(
        Xe, WkT, nullptr, nullptr, Kb, MPAD_E, CIN, CENC, NB * SKV);
    // V = Xe @ Wv -> fp32
    gemm_bf16_kernel<<<dim3(CIN / 256, MPAD_E / 64), 256, 0, stream>>>(
        Xe, WvT, nullptr, Vf, nullptr, MPAD_E, CIN, CENC, NB * SKV);

    // CORA erase -> Vt (transposed per head, kv padded)
    cora_kernel<<<dim3(NB * SKV_P), 256, 0, stream>>>(Vf, Bpad, uhat, ahat, Vt);

    // fused attention -> Attn (bf16)
    attention_kernel<<<dim3(SQ / 64, NB * NHEAD), 128, 0, stream>>>(Qb, Kb, Vt, Attn);

    // out = Attn @ Wo + bo -> fp32
    gemm_bf16_kernel<<<dim3(CIN / 256, (NB * SQ) / 64), 256, 0, stream>>>(
        Attn, WoT, bo, out, nullptr, NB * SQ, CIN, CIN, NB * SQ);
}
